// SDPAttention_55027120996798
// MI455X (gfx1250) — matrix operand order repaired and checked by reading, not compiled, not run on hardware
//
#include <hip/hip_runtime.h>
#include <hip/hip_bf16.h>
#include <stdint.h>

// Problem constants (match reference)
#define B_  2
#define L_  1024
#define D_  1024
#define H_  16
#define HD_ 64
#define M_  (B_ * L_)   // 2048 token rows

typedef __attribute__((ext_vector_type(8)))  _Float16 h8;
typedef __attribute__((ext_vector_type(16))) _Float16 v16h;
typedef __attribute__((ext_vector_type(8)))  float    v8f;

union FragU { v16h v; h8 h[2]; };

// ---------------------------------------------------------------------------
// WMMA fragment loaders (wave32, 16x16x32 f16 shapes per CDNA5 ISA)
//
// A (16x32, MxK): lane 0-15 -> row = lane, halves K = k0+{0..7} then k0+{16..23}
//                 lane 16-31 -> row = lane-16, halves K = k0+{8..15}, k0+{24..31}
// B (32x16, KxN): lane holds column n = lane&15; same K order as A:
//                 lanes 0-15 K = k0+{0..7}, k0+{16..23}; lanes 16-31
//                 K = k0+{8..15}, k0+{24..31} (source rows contiguous in K).
// ---------------------------------------------------------------------------
__device__ __forceinline__ v16h frag_a_global(const _Float16* base, int row0,
                                              int rs, int k0, int lane) {
  const int half = lane >> 4;
  const _Float16* p = base + (size_t)(row0 + (lane & 15)) * rs + (k0 + half * 8);
  FragU f;
  f.h[0] = *(const h8*)(p);
  f.h[1] = *(const h8*)(p + 16);
  return f.v;
}

__device__ __forceinline__ v16h frag_b_global(const _Float16* base, int n0,
                                              int rs, int k0, int lane) {
  const int half = lane >> 4;
  const _Float16* p = base + (size_t)(n0 + (lane & 15)) * rs + (k0 + half * 8);
  FragU f;
  f.h[0] = *(const h8*)(p);
  f.h[1] = *(const h8*)(p + 16);
  return f.v;
}

__device__ __forceinline__ v16h frag_a_lds(const _Float16* rowbase, int rs,
                                           int k0, int lane) {
  const int half = lane >> 4;
  const _Float16* p = rowbase + (size_t)(lane & 15) * rs + (k0 + half * 8);
  FragU f;
  f.h[0] = *(const h8*)(p);
  f.h[1] = *(const h8*)(p + 16);
  return f.v;
}

__device__ __forceinline__ v8f wmma16(v16h a, v16h b, v8f c) {
  return __builtin_amdgcn_wmma_f32_16x16x32_f16(false, a, false, b,
                                                (short)0, c, false, false);
}

// ---------------------------------------------------------------------------
// fp32 -> f16 conversion
// ---------------------------------------------------------------------------
__global__ void cvt_f32_f16(const float* __restrict__ src,
                            _Float16* __restrict__ dst, int n) {
  int i = blockIdx.x * blockDim.x + threadIdx.x;
  int stride = gridDim.x * blockDim.x;
  for (; i < n; i += stride) dst[i] = (_Float16)src[i];
}

// ---------------------------------------------------------------------------
// Y = X @ W^T + bias      X:[M_,D_] f16 row-major, W:[D_,D_] f16 row-major
// MODE 0: write f16, head-split layout  [B,H,L,HD]   (Q, K projections)
// MODE 1: write f16, transposed heads   [B,H,HD,L]   (V projection -> Vt)
// MODE 2: write f32, flat [M_,D_] to d_out (+bias)   (output projection)
// Grid (M_/128, D_/64), 128 threads (4 waves); each wave: 32 rows x 64 cols
// (2 A-frags reuse each of the 4 B-frags -> 8 WMMAs per 12 b128 loads).
// K-loop unrolled 4x: 32 WMMAs / 48 loads per body for deep load clauses.
// ---------------------------------------------------------------------------
template <int MODE>
__global__ __launch_bounds__(128) void gemm_wmma(
    const _Float16* __restrict__ A, const _Float16* __restrict__ W,
    const float* __restrict__ bias, void* __restrict__ out) {
  const int lane = threadIdx.x & 31;
  const int wave = threadIdx.x >> 5;
  const int row0 = blockIdx.x * 128 + wave * 32;
  const int col0 = blockIdx.y * 64;

  v8f acc[2][4] = {};

#pragma unroll 4
  for (int k0 = 0; k0 < D_; k0 += 32) {
    v16h a0 = frag_a_global(A, row0, D_, k0, lane);
    v16h a1 = frag_a_global(A, row0 + 16, D_, k0, lane);
#pragma unroll
    for (int t = 0; t < 4; ++t) {
      v16h b = frag_b_global(W, col0 + t * 16, D_, k0, lane);
      acc[0][t] = wmma16(a0, b, acc[0][t]);
      acc[1][t] = wmma16(a1, b, acc[1][t]);
    }
  }

  const int half = lane >> 4;
  const int cb = lane & 15;
#pragma unroll
  for (int r = 0; r < 2; ++r) {
#pragma unroll
    for (int t = 0; t < 4; ++t) {
      const int col = col0 + t * 16 + cb;
      const float bb = bias[col];
#pragma unroll
      for (int j = 0; j < 8; ++j) {
        const int row = row0 + r * 16 + j + half * 8;  // C layout rows j / j+8
        const float val = acc[r][t][j] + bb;
        if (MODE == 2) {
          ((float*)out)[(size_t)row * D_ + col] = val;
        } else {
          const int bidx = row >> 10, l = row & (L_ - 1);
          const int hh = col >> 6, d = col & (HD_ - 1);
          _Float16* o = (_Float16*)out;
          if (MODE == 0)
            o[(((size_t)(bidx * H_ + hh)) * L_ + l) * HD_ + d] = (_Float16)val;
          else
            o[(((size_t)(bidx * H_ + hh)) * HD_ + d) * L_ + l] = (_Float16)val;
        }
      }
    }
  }
}

// ---------------------------------------------------------------------------
// Flash attention, causal + padding mask, online softmax.
// Grid: B_*H_*(L_/64) WGs of 128 threads. Each WG: one (b,h), 64 query rows;
// each wave owns 16 query rows. P routed through per-wave LDS to convert the
// C-layout scores into A-layout fragments for the PV WMMAs.
// ---------------------------------------------------------------------------
#define NEG_INF (-3.0e38f)

__global__ __launch_bounds__(128) void flash_attn(
    const _Float16* __restrict__ Qp, const _Float16* __restrict__ Kp,
    const _Float16* __restrict__ Vt, const uint8_t* __restrict__ pmask,
    _Float16* __restrict__ AO) {
  __shared__ _Float16 ldsP[4][16][80];  // stride 80 halves = 160B (16B aligned)

  const int lane = threadIdx.x & 31;
  const int wave = threadIdx.x >> 5;
  const int qb = blockIdx.x & (L_ / 64 - 1);
  const int bh = blockIdx.x >> 4;
  const int b = bh >> 4;
  const int h = bh & (H_ - 1);

  const _Float16* Q = Qp + (size_t)bh * L_ * HD_;
  const _Float16* K = Kp + (size_t)bh * L_ * HD_;
  const _Float16* V = Vt + (size_t)bh * HD_ * L_;  // [HD_][L_]
  const uint8_t* pm = pmask + (size_t)b * L_;

  const int qrb = qb * 64 + wave * 16;
  const int half = lane >> 4;
  const int cb = lane & 15;

  const v16h q0 = frag_a_global(Q, qrb, HD_, 0, lane);
  const v16h q1 = frag_a_global(Q, qrb, HD_, 32, lane);

  int qrow[8];
  bool qv[8];
#pragma unroll
  for (int j = 0; j < 8; ++j) {
    qrow[j] = qrb + j + half * 8;
    qv[j] = pm[qrow[j]] != 0;
  }

  float m_run[8], l_run[8];
#pragma unroll
  for (int j = 0; j < 8; ++j) { m_run[j] = NEG_INF; l_run[j] = 0.0f; }
  v8f o[4] = {};
  const float scale = 0.125f;  // 1/sqrt(64)

  const int nkb = qb + 1;  // causal: only blocks at/below the diagonal
  for (int kb = 0; kb < nkb; ++kb) {
    const int kv0 = kb * 64;

    // S = Q K^T : 4 tiles of 16q x 16k, K-dim = head dim (2 WMMAs each)
    v8f s[4];
#pragma unroll
    for (int t = 0; t < 4; ++t) {
      v16h b0 = frag_b_global(K, kv0 + t * 16, HD_, 0, lane);
      v16h b1 = frag_b_global(K, kv0 + t * 16, HD_, 32, lane);
      v8f z = {};
      z = wmma16(q0, b0, z);
      z = wmma16(q1, b1, z);
      s[t] = z;
    }

    // scale + causal/padding mask, per-lane row max
    float mt[8];
#pragma unroll
    for (int j = 0; j < 8; ++j) mt[j] = NEG_INF;
#pragma unroll
    for (int t = 0; t < 4; ++t) {
      const int kcol = kv0 + t * 16 + cb;
      const bool kvld = pm[kcol] != 0;
#pragma unroll
      for (int j = 0; j < 8; ++j) {
        float sv = s[t][j] * scale;
        const bool keep = kvld && qv[j] && (kcol <= qrow[j]);
        sv = keep ? sv : NEG_INF;
        s[t][j] = sv;
        mt[j] = fmaxf(mt[j], sv);
      }
    }

    // reduce row max across the 16 lanes of each half, online rescale
#pragma unroll
    for (int j = 0; j < 8; ++j) {
#pragma unroll
      for (int off = 8; off >= 1; off >>= 1)
        mt[j] = fmaxf(mt[j], __shfl_xor(mt[j], off, 32));
      const float mn = fmaxf(m_run[j], mt[j]);
      const float alpha = __expf(m_run[j] - mn);
      m_run[j] = mn;
      l_run[j] *= alpha;
#pragma unroll
      for (int t = 0; t < 4; ++t) o[t][j] *= alpha;
    }

    // P = exp(S - m), row sums, stage P in LDS as f16 (C-layout -> A-layout)
    float rs[8];
#pragma unroll
    for (int j = 0; j < 8; ++j) rs[j] = 0.0f;
#pragma unroll
    for (int t = 0; t < 4; ++t) {
#pragma unroll
      for (int j = 0; j < 8; ++j) {
        const float p = __expf(s[t][j] - m_run[j]);
        rs[j] += p;
        ldsP[wave][j + half * 8][t * 16 + cb] = (_Float16)p;
      }
    }
#pragma unroll
    for (int j = 0; j < 8; ++j) {
#pragma unroll
      for (int off = 8; off >= 1; off >>= 1)
        rs[j] += __shfl_xor(rs[j], off, 32);
      l_run[j] += rs[j];
    }
    __syncthreads();

    // O += P V : A = P (from LDS), B = Vt rows (d) contiguous in l
    const v16h p0 = frag_a_lds(&ldsP[wave][0][0], 80, 0, lane);
    const v16h p1 = frag_a_lds(&ldsP[wave][0][0], 80, 32, lane);
#pragma unroll
    for (int t = 0; t < 4; ++t) {
      v16h vb0 = frag_b_global(V, t * 16, L_, kv0, lane);
      v16h vb1 = frag_b_global(V, t * 16, L_, kv0 + 32, lane);
      o[t] = wmma16(p0, vb0, o[t]);
      o[t] = wmma16(p1, vb1, o[t]);
    }
    __syncthreads();
  }

  // normalize and write concat-head f16 output [B, L, H*HD]
#pragma unroll
  for (int t = 0; t < 4; ++t) {
#pragma unroll
    for (int j = 0; j < 8; ++j) {
      const float val = o[t][j] / l_run[j];
      AO[((size_t)(b * L_ + qrow[j])) * D_ + h * HD_ + t * 16 + cb] =
          (_Float16)val;
    }
  }
}

// ---------------------------------------------------------------------------
extern "C" void kernel_launch(void* const* d_in, const int* in_sizes, int n_in,
                              void* d_out, int out_size, void* d_ws,
                              size_t ws_size, hipStream_t stream) {
  (void)in_sizes; (void)n_in; (void)out_size; (void)ws_size;
  const float* q = (const float*)d_in[0];
  const float* k = (const float*)d_in[1];
  const float* v = (const float*)d_in[2];
  const uint8_t* mask = (const uint8_t*)d_in[3];  // bool [B,L]
  const float* Wq = (const float*)d_in[4];
  const float* bq = (const float*)d_in[5];
  const float* Wk = (const float*)d_in[6];
  const float* bk = (const float*)d_in[7];
  const float* Wv = (const float*)d_in[8];
  const float* bv = (const float*)d_in[9];
  const float* Wo = (const float*)d_in[10];
  const float* bo = (const float*)d_in[11];

  char* ws = (char*)d_ws;
  const size_t MB = (size_t)1 << 20;
  _Float16* qh  = (_Float16*)(ws + 0 * MB);   // 4 MiB each (2M f16)
  _Float16* kh  = (_Float16*)(ws + 4 * MB);
  _Float16* vh  = (_Float16*)(ws + 8 * MB);
  _Float16* wqh = (_Float16*)(ws + 12 * MB);  // 2 MiB each (1M f16)
  _Float16* wkh = (_Float16*)(ws + 14 * MB);
  _Float16* wvh = (_Float16*)(ws + 16 * MB);
  _Float16* woh = (_Float16*)(ws + 18 * MB);
  _Float16* Qp  = (_Float16*)(ws + 20 * MB);  // [B,H,L,HD] f16
  _Float16* Kp  = (_Float16*)(ws + 24 * MB);  // [B,H,L,HD] f16
  _Float16* Vt  = (_Float16*)(ws + 28 * MB);  // [B,H,HD,L] f16
  _Float16* AO  = (_Float16*)(ws + 32 * MB);  // [M_,D_] f16

  const int NX = B_ * L_ * D_;  // 2M
  const int NW = D_ * D_;       // 1M
  cvt_f32_f16<<<2048, 256, 0, stream>>>(q, qh, NX);
  cvt_f32_f16<<<2048, 256, 0, stream>>>(k, kh, NX);
  cvt_f32_f16<<<2048, 256, 0, stream>>>(v, vh, NX);
  cvt_f32_f16<<<1024, 256, 0, stream>>>(Wq, wqh, NW);
  cvt_f32_f16<<<1024, 256, 0, stream>>>(Wk, wkh, NW);
  cvt_f32_f16<<<1024, 256, 0, stream>>>(Wv, wvh, NW);
  cvt_f32_f16<<<1024, 256, 0, stream>>>(Wo, woh, NW);

  dim3 grid(M_ / 128, D_ / 64), blk(128);
  gemm_wmma<0><<<grid, blk, 0, stream>>>(qh, wqh, bq, Qp);
  gemm_wmma<0><<<grid, blk, 0, stream>>>(kh, wkh, bk, Kp);
  gemm_wmma<1><<<grid, blk, 0, stream>>>(vh, wvh, bv, Vt);

  flash_attn<<<B_ * H_ * (L_ / 64), 128, 0, stream>>>(Qp, Kp, Vt, mask, AO);

  gemm_wmma<2><<<grid, blk, 0, stream>>>(AO, woh, bo, d_out);
}
